// EdgeModel_out_24756191494618
// MI455X (gfx1250) — matrix operand order repaired and checked by reading, not compiled, not run on hardware
//
#include <hip/hip_runtime.h>
#include <hip/hip_bf16.h>

typedef __attribute__((ext_vector_type(16))) __bf16 v16bf;
typedef __attribute__((ext_vector_type(8)))  float  v8f;

#define WAVES_PER_WG 8
#define THREADS (WAVES_PER_WG * 32)

// Fused edge-MLP:
//   h = concat(x_s[src], x_t[tgt], edge_attr, u[batch_e])  (per edge, 256 wide)
//   h = LeakyReLU_0.1(h @ W1 + b1)   (256 -> 64)
//   o = h @ W2 + b2                  (64 -> 64)
// One wave32 computes a 16-edge x 64-out tile with v_wmma_f32_16x16x32_bf16.
__global__ __launch_bounds__(THREADS)
void edge_mlp_wmma_kernel(const float* __restrict__ xs,  const float* __restrict__ xt,
                          const int*   __restrict__ ei,  const float* __restrict__ ea,
                          const float* __restrict__ u,   const int*   __restrict__ be,
                          const float* __restrict__ W1,  const float* __restrict__ b1,
                          const float* __restrict__ W2,  const float* __restrict__ b2,
                          float* __restrict__ out, int n_edges)
{
    // B-fragment-swizzled bf16 weights:
    //   fragment f = kt*4+nt, lane l holds column n = nt*16+(l&15),
    //   K = kt*32 + (l>>4)*8 + j + (j&8), j = 0..15 (two ds_load_b128 per use).
    __shared__ __bf16 ldsW1[8 * 4 * 32 * 16];        // 16384 bf16 = 32 KB
    __shared__ __bf16 ldsW2[2 * 4 * 32 * 16];        //  4096 bf16 =  8 KB
    __shared__ __bf16 ldsH [WAVES_PER_WG * 1024];    // per-wave 16x64 A-frag staging, 16 KB

    const int tid = threadIdx.x;

    // ---- Stage W1 (256x64) into LDS, bf16, fragment order ----
    for (int slot = tid; slot < 8 * 4 * 32; slot += THREADS) {
        const int f  = slot >> 5, l = slot & 31;
        const int kt = f >> 2,  nt = f & 3;
        const int kb = kt * 32 + (l >> 4) * 8;
        const int n  = nt * 16 + (l & 15);
        #pragma unroll
        for (int j = 0; j < 16; ++j)
            ldsW1[slot * 16 + j] = (__bf16)W1[(kb + j + (j & 8)) * 64 + n];
    }
    // ---- Stage W2 (64x64) ----
    for (int slot = tid; slot < 2 * 4 * 32; slot += THREADS) {
        const int f  = slot >> 5, l = slot & 31;
        const int kt = f >> 2,  nt = f & 3;
        const int kb = kt * 32 + (l >> 4) * 8;
        const int n  = nt * 16 + (l & 15);
        #pragma unroll
        for (int j = 0; j < 16; ++j)
            ldsW2[slot * 16 + j] = (__bf16)W2[(kb + j + (j & 8)) * 64 + n];
    }
    __syncthreads();

    const int lane   = tid & 31;
    const int wave   = tid >> 5;
    const int laneLo = lane & 15;
    const int laneHi = lane >> 4;
    __bf16* hstg = &ldsH[wave * 1024];

    // Per-lane bias values for C layout (lane -> column N = nt*16 + laneLo).
    float b1v[4], b2v[4];
    #pragma unroll
    for (int nt = 0; nt < 4; ++nt) {
        b1v[nt] = b1[nt * 16 + laneLo];
        b2v[nt] = b2[nt * 16 + laneLo];
    }

    const int nTiles     = (n_edges + 15) >> 4;
    const int waveGlobal = blockIdx.x * WAVES_PER_WG + wave;
    const int waveStride = gridDim.x * WAVES_PER_WG;

    for (int tile = waveGlobal; tile < nTiles; tile += waveStride) {
        const int base = tile * 16;
        int e = base + laneLo;
        if (e >= n_edges) e = n_edges - 1;          // clamp (tail safety)

        const long srcI = ei[e];
        const long tgtI = ei[n_edges + e];
        const long bI   = be[e];
        const float* seg[4] = { xs + srcI * 64, xt + tgtI * 64,
                                ea + (long)e * 64, u + bI * 64 };

        v8f acc[4] = {};    // 16x64 f32 accumulators (4 C tiles)

        // ---- GEMM1: h(16x256) @ W1(256x64), K in 8 steps of 32 ----
        #pragma unroll
        for (int kt = 0; kt < 8; ++kt) {
            // A layout: lane row = laneLo; chunks K = kt*32 + laneHi*8 + {0..7, 16..23}
            const float* p = seg[kt >> 1] + (kt & 1) * 32 + laneHi * 8;
            v16bf a;
            #pragma unroll
            for (int j = 0; j < 8; ++j) {
                a[j]     = (__bf16)p[j];
                a[8 + j] = (__bf16)p[16 + j];
            }
            #pragma unroll
            for (int nt = 0; nt < 4; ++nt) {
                const v16bf bf = *(const v16bf*)&ldsW1[((kt * 4 + nt) * 32 + lane) * 16];
                acc[nt] = __builtin_amdgcn_wmma_f32_16x16x32_bf16(
                    false, a, false, bf, (short)0, acc[nt], false, false);
            }
        }

        // ---- bias + LeakyReLU(0.1), restage C layout -> A-fragment layout in LDS ----
        #pragma unroll
        for (int nt = 0; nt < 4; ++nt) {
            const int j = (laneLo & 7) + (nt & 1) * 8;
            #pragma unroll
            for (int r = 0; r < 8; ++r) {
                float hv = acc[nt][r] + b1v[nt];
                hv = fmaxf(hv, 0.1f * hv);
                const int m  = r + laneHi * 8;
                const int lp = m + ((laneLo >> 3) & 1) * 16;
                hstg[(nt >> 1) * 512 + lp * 16 + j] = (__bf16)hv;
            }
        }

        // ---- GEMM2: h1(16x64) @ W2(64x64), K in 2 steps of 32 ----
        v8f acc2[4] = {};
        #pragma unroll
        for (int kt = 0; kt < 2; ++kt) {
            const v16bf a2 = *(const v16bf*)&hstg[kt * 512 + lane * 16];
            #pragma unroll
            for (int nt = 0; nt < 4; ++nt) {
                const v16bf bf = *(const v16bf*)&ldsW2[((kt * 4 + nt) * 32 + lane) * 16];
                acc2[nt] = __builtin_amdgcn_wmma_f32_16x16x32_bf16(
                    false, a2, false, bf, (short)0, acc2[nt], false, false);
            }
        }

        // ---- bias + store fp32 output (coalesced across lanes) ----
        #pragma unroll
        for (int nt = 0; nt < 4; ++nt) {
            #pragma unroll
            for (int r = 0; r < 8; ++r) {
                const int m   = r + laneHi * 8;
                const int row = base + m;
                if (row < n_edges)
                    out[(size_t)row * 64 + nt * 16 + laneLo] = acc2[nt][r] + b2v[nt];
            }
        }
    }
}

extern "C" void kernel_launch(void* const* d_in, const int* in_sizes, int n_in,
                              void* d_out, int out_size, void* d_ws, size_t ws_size,
                              hipStream_t stream) {
    const float* xs = (const float*)d_in[0];
    const float* xt = (const float*)d_in[1];
    const int*   ei = (const int*)  d_in[2];
    const float* ea = (const float*)d_in[3];
    const float* u  = (const float*)d_in[4];
    const int*   be = (const int*)  d_in[5];
    const float* W1 = (const float*)d_in[6];
    const float* b1 = (const float*)d_in[7];
    const float* W2 = (const float*)d_in[8];
    const float* b2 = (const float*)d_in[9];
    float* out = (float*)d_out;

    const int n_edges = in_sizes[5];            // batch_e has one entry per edge
    const int nTiles  = (n_edges + 15) >> 4;
    int blocks = (nTiles + 8 * WAVES_PER_WG - 1) / (8 * WAVES_PER_WG); // ~8 tiles/wave
    if (blocks > 1024) blocks = 1024;
    if (blocks < 1)    blocks = 1;

    edge_mlp_wmma_kernel<<<blocks, THREADS, 0, stream>>>(
        xs, xt, ei, ea, u, be, W1, b1, W2, b2, out, n_edges);
}
